// Decoder_24781961298112
// MI455X (gfx1250) — matrix operand order repaired and checked by reading, not compiled, not run on hardware
//
#include <hip/hip_runtime.h>
#include <hip/hip_bf16.h>

typedef __attribute__((ext_vector_type(16))) __bf16 v16bf;
typedef __attribute__((ext_vector_type(8)))  __bf16 v8bf;
typedef __attribute__((ext_vector_type(8)))  float  v8f;

#define BB 64
#define TT 50
#define DD 512
#define HH 1024
#define VV 32000

// ---------- helpers ----------

__device__ __forceinline__ unsigned short f2bf(float f) {
    unsigned int u = __float_as_uint(f);
    unsigned int r = u + 0x7fffu + ((u >> 16) & 1u);   // RNE
    return (unsigned short)(r >> 16);
}

__device__ __forceinline__ float sigmoidf_(float x) {
    return 1.0f / (1.0f + __expf(-x));
}

// monotone float -> u32 key
__device__ __forceinline__ unsigned int ordered_key(float f) {
    unsigned int u = __float_as_uint(f);
    return (u & 0x80000000u) ? ~u : (u | 0x80000000u);
}

__device__ __forceinline__ unsigned int hash3(unsigned int t, unsigned int r, unsigned int c) {
    unsigned int x = t * 0x9E3779B1u ^ r * 0x85EBCA77u ^ c * 0xC2B2AE3Du;
    x ^= x >> 16; x *= 0x7feb352du;
    x ^= x >> 15; x *= 0x846ca68bu;
    x ^= x >> 16;
    return x;
}

// A tile (16x32 bf16, row-major source [rows, ldk]) in this code's contiguous k order (B matches):
// lane l: row = m0 + (l&15); elems 0..15 = K[k0 + hg*16 .. k0 + hg*16 + 15] contiguous
__device__ __forceinline__ v16bf load_a_tile(const unsigned short* base, int ldk,
                                             int row, int k0, int hg) {
    const v8bf* p0 = (const v8bf*)(base + (size_t)row * ldk + k0 + hg * 16);
    const v8bf* p1 = (const v8bf*)(base + (size_t)row * ldk + k0 + hg * 16 + 8);
    v8bf lo = *p0, hi = *p1;
    v16bf a;
#pragma unroll
    for (int i = 0; i < 8; ++i) { a[i] = lo[i]; a[i + 8] = hi[i]; }
    return a;
}

// B tile (32x16 bf16) from row-major weight W[N,K]: column n of B = row n of W.
// lane l: col = l&15, K = hg*16 .. hg*16+15 contiguous -> one 32B load.
__device__ __forceinline__ v16bf load_b_tile(const unsigned short* w, int ldk,
                                             int nrow, int k0, int hg) {
    return *(const v16bf*)(w + (size_t)nrow * ldk + k0 + hg * 16);
}

__device__ __forceinline__ v8f wmma_bf16(v16bf a, v16bf b, v8f c) {
    return __builtin_amdgcn_wmma_f32_16x16x32_bf16(false, a, false, b, (short)0, c, false, false);
}

// ---------- kernels ----------

__global__ void k_convert_bf16(const float* __restrict__ src,
                               unsigned short* __restrict__ dst, long long n) {
    long long i = (long long)blockIdx.x * blockDim.x + threadIdx.x;
    long long stride = (long long)gridDim.x * blockDim.x;
    for (; i < n; i += stride) dst[i] = f2bf(src[i]);
}

__global__ void k_init(const float* __restrict__ seq_emb,   // [B,H]
                       const float* __restrict__ emb_table, // [V,D]
                       const int* __restrict__ eos_ptr,
                       float* __restrict__ hA, unsigned short* __restrict__ hbA,
                       unsigned short* __restrict__ x_bf) {
    int i = blockIdx.x * blockDim.x + threadIdx.x;
    if (i < BB * HH) {
        float v = seq_emb[i];
        hA[i]  = v;
        hbA[i] = f2bf(v);
    }
    if (i < BB * DD) {
        int eos = eos_ptr[0];
        int c = i & (DD - 1);
        x_bf[i] = f2bf(emb_table[(size_t)eos * DD + c]);
    }
}

// GRU gates + cell math. grid = H/16 blocks, 128 threads (4 waves; wave w owns rows 16w..16w+15).
__global__ void k_gru_gates(const unsigned short* __restrict__ x_bf,    // [B,D]
                            const unsigned short* __restrict__ h_bf_in, // [B,H]
                            const float*          __restrict__ h_in,    // [B,H]
                            const unsigned short* __restrict__ Wih,     // [3H,D] bf16
                            const unsigned short* __restrict__ Whh,     // [3H,H] bf16
                            const float* __restrict__ b_ih,
                            const float* __restrict__ b_hh,
                            float*          __restrict__ h_out,
                            unsigned short* __restrict__ h_bf_out,
                            unsigned long long* __restrict__ slots) {
    // reset argmax slots for this step (runs before the logits kernel in stream order)
    if (blockIdx.x == 0 && threadIdx.x < BB) slots[threadIdx.x] = 0ull;

    const int lane = threadIdx.x & 31;
    const int wave = threadIdx.x >> 5;
    const int n    = lane & 15;
    const int hg   = lane >> 4;
    const int j0   = blockIdx.x * 16;  // column tile within H
    const int m0   = wave * 16;        // row tile within B

    v8f acc[6] = {};

    // gi = x @ W_ih^T   (K = D)
    for (int k0 = 0; k0 < DD; k0 += 32) {
        v16bf a = load_a_tile(x_bf, DD, m0 + n, k0, hg);
#pragma unroll
        for (int g = 0; g < 3; ++g) {
            v16bf b = load_b_tile(Wih, DD, g * HH + j0 + n, k0, hg);
            acc[g] = wmma_bf16(a, b, acc[g]);
        }
    }
    // gh = h @ W_hh^T   (K = H)
    for (int k0 = 0; k0 < HH; k0 += 32) {
        v16bf a = load_a_tile(h_bf_in, HH, m0 + n, k0, hg);
#pragma unroll
        for (int g = 0; g < 3; ++g) {
            v16bf b = load_b_tile(Whh, HH, g * HH + j0 + n, k0, hg);
            acc[3 + g] = wmma_bf16(a, b, acc[3 + g]);
        }
    }

    const int col = j0 + n;
    const float bir = b_ih[col],           bhr = b_hh[col];
    const float biz = b_ih[HH + col],      bhz = b_hh[HH + col];
    const float bin = b_ih[2 * HH + col],  bhn = b_hh[2 * HH + col];

#pragma unroll
    for (int r = 0; r < 8; ++r) {
        int row = m0 + hg * 8 + r;
        float ir = acc[0][r] + bir, iz = acc[1][r] + biz, in_ = acc[2][r] + bin;
        float hr = acc[3][r] + bhr, hz = acc[4][r] + bhz, hn  = acc[5][r] + bhn;
        float rg = sigmoidf_(ir + hr);
        float z  = sigmoidf_(iz + hz);
        float nn = tanhf(in_ + rg * hn);
        float ho = h_in[(size_t)row * HH + col];
        float hv = (1.0f - z) * nn + z * ho;
        h_out[(size_t)row * HH + col]    = hv;
        h_bf_out[(size_t)row * HH + col] = f2bf(hv);
    }
}

// logits = h @ fc_w^T + fc_b, fused Gumbel-argmax. grid = V/64 blocks, 128 threads.
__global__ void k_logits(const unsigned short* __restrict__ h_bf,  // [B,H]
                         const unsigned short* __restrict__ fcw,   // [V,H] bf16
                         const float* __restrict__ fcb,            // [V]
                         float* __restrict__ out_logits,           // [T,B,V] base
                         unsigned long long* __restrict__ slots,   // [B]
                         int t) {
    const int lane = threadIdx.x & 31;
    const int wave = threadIdx.x >> 5;
    const int n    = lane & 15;
    const int hg   = lane >> 4;
    const int m0   = wave * 16;
    const int nb   = blockIdx.x * 64;

    v8f acc[4] = {};
    for (int k0 = 0; k0 < HH; k0 += 32) {
        v16bf a = load_a_tile(h_bf, HH, m0 + n, k0, hg);
#pragma unroll
        for (int tn = 0; tn < 4; ++tn) {
            v16bf b = load_b_tile(fcw, HH, nb + tn * 16 + n, k0, hg);
            acc[tn] = wmma_bf16(a, b, acc[tn]);
        }
    }

    float bias[4];
#pragma unroll
    for (int tn = 0; tn < 4; ++tn) bias[tn] = fcb[nb + tn * 16 + n];

    float* Lt = out_logits + (size_t)t * BB * VV;
#pragma unroll
    for (int r = 0; r < 8; ++r) {
        int row = m0 + hg * 8 + r;
        unsigned long long best = 0ull;
#pragma unroll
        for (int tn = 0; tn < 4; ++tn) {
            int c = nb + tn * 16 + n;
            float v = acc[tn][r] + bias[tn];
            Lt[(size_t)row * VV + c] = v;
            // Gumbel noise, deterministic in (t,row,c)
            unsigned int h = hash3((unsigned)t, (unsigned)row, (unsigned)c);
            float u = ((float)h + 0.5f) * (1.0f / 4294967296.0f);
            float g = v - __logf(-__logf(u));
            unsigned long long cand =
                ((unsigned long long)ordered_key(g) << 32) | (unsigned int)c;
            best = best > cand ? best : cand;
        }
        atomicMax(&slots[row], best);
    }
}

// pick winner, emit index, gather next-step embedding. grid = B blocks, 128 threads.
__global__ void k_sample_gather(const unsigned long long* __restrict__ slots,
                                const unsigned short* __restrict__ emb_bf, // [V,D]
                                int* __restrict__ out_idx,                  // [T,B]
                                unsigned short* __restrict__ x_bf,          // [B,D]
                                int t) {
    int b = blockIdx.x;
    unsigned long long s = slots[b];
    int idx = (int)(s & 0xffffffffu);
    if (threadIdx.x == 0) out_idx[(size_t)t * BB + b] = idx;
    const uint2* src = (const uint2*)(emb_bf + (size_t)idx * DD);
    uint2*       dst = (uint2*)(x_bf + (size_t)b * DD);
    dst[threadIdx.x] = src[threadIdx.x];   // 128 threads * 8B = 1024B = 512 bf16
}

// ---------- launcher ----------

extern "C" void kernel_launch(void* const* d_in, const int* in_sizes, int n_in,
                              void* d_out, int out_size, void* d_ws, size_t ws_size,
                              hipStream_t stream) {
    const float* seq_emb = (const float*)d_in[0]; // [B,H]
    const float* emb     = (const float*)d_in[1]; // [V,D]
    const float* W_ih    = (const float*)d_in[2]; // [3H,D]
    const float* W_hh    = (const float*)d_in[3]; // [3H,H]
    const float* b_ih    = (const float*)d_in[4];
    const float* b_hh    = (const float*)d_in[5];
    const float* fc_w    = (const float*)d_in[6]; // [V,H]
    const float* fc_b    = (const float*)d_in[7];
    const int*   eos     = (const int*)d_in[8];

    int*   out_idx    = (int*)d_out;               // [T,B] int32
    float* out_logits = (float*)d_out + TT * BB;   // [T,B,V] f32

    // workspace carve-up (256B aligned)
    char* ws = (char*)d_ws;
    size_t off = 0;
    auto carve = [&](size_t bytes) -> char* {
        char* p = ws + off;
        off = (off + bytes + 255) & ~(size_t)255;
        return p;
    };
    unsigned short* Wih_bf = (unsigned short*)carve((size_t)3 * HH * DD * 2);
    unsigned short* Whh_bf = (unsigned short*)carve((size_t)3 * HH * HH * 2);
    unsigned short* fcw_bf = (unsigned short*)carve((size_t)VV * HH * 2);
    unsigned short* emb_bf = (unsigned short*)carve((size_t)VV * DD * 2);
    float*          hA     = (float*)carve((size_t)BB * HH * 4);
    float*          hB     = (float*)carve((size_t)BB * HH * 4);
    unsigned short* hbA    = (unsigned short*)carve((size_t)BB * HH * 2);
    unsigned short* hbB    = (unsigned short*)carve((size_t)BB * HH * 2);
    unsigned short* x_bf   = (unsigned short*)carve((size_t)BB * DD * 2);
    unsigned long long* slots = (unsigned long long*)carve((size_t)BB * 8);
    (void)ws_size; (void)in_sizes; (void)n_in; (void)out_size;

    // 1) bf16 weight conversion (once per launch; deterministic)
    k_convert_bf16<<<1024, 256, 0, stream>>>(W_ih, Wih_bf, (long long)3 * HH * DD);
    k_convert_bf16<<<1024, 256, 0, stream>>>(W_hh, Whh_bf, (long long)3 * HH * HH);
    k_convert_bf16<<<2048, 256, 0, stream>>>(fc_w, fcw_bf, (long long)VV * HH);
    k_convert_bf16<<<2048, 256, 0, stream>>>(emb,  emb_bf, (long long)VV * DD);

    // 2) h0 / x0
    k_init<<<(BB * HH + 255) / 256, 256, 0, stream>>>(seq_emb, emb, eos, hA, hbA, x_bf);

    // 3) decode loop (ping-pong h buffers)
    for (int t = 0; t < TT; ++t) {
        const float*          h_in    = (t & 1) ? hB  : hA;
        const unsigned short* h_bf_in = (t & 1) ? hbB : hbA;
        float*                h_out   = (t & 1) ? hA  : hB;
        unsigned short*       h_bf_out= (t & 1) ? hbA : hbB;

        k_gru_gates<<<HH / 16, 128, 0, stream>>>(x_bf, h_bf_in, h_in,
                                                 Wih_bf, Whh_bf, b_ih, b_hh,
                                                 h_out, h_bf_out, slots);
        k_logits<<<VV / 64, 128, 0, stream>>>(h_bf_out, fcw_bf, fc_b,
                                              out_logits, slots, t);
        k_sample_gather<<<BB, 128, 0, stream>>>(slots, emb_bf, out_idx, x_bf, t);
    }
}
